// lpLSTM_76879914598626
// MI455X (gfx1250) — matrix operand order repaired and checked by reading, not compiled, not run on hardware
//
#include <hip/hip_runtime.h>
#include <hip/hip_bf16.h>

// lpLSTM on MI455X (gfx1250): bf16 WMMA GEMMs + persistent fused recurrent scan.
//
// T=512, B=32, IN=1024, H=1024, 4H=4096.
//
// Phase 1: convert x / W_ih / W_hh / h0 to bf16 in ws, zero barrier state.
// Phase 2: x_proj = x_bf16 @ W_ih^T (+ biases), WMMA tiled, stored TRANSPOSED
//          as [T, 4H, B] so the scan reads it with coalesced float4 loads.
// Phase 3: persistent scan, 32 blocks x 256 threads = 256 waves.
//          Wave pair (kpart=0/1) splits K=1024; each owns a 16-row x 16-hidden
//          slice and computes all four gate tiles (i,f,g,o) for it with
//          v_wmma_f32_16x16x32_bf16, partials combined through LDS.
//          c and h(prev) live in registers across all 512 steps; h is
//          republished as bf16 (double-buffered) + one grid barrier per step.

#define T_DIM 512
#define B_DIM 32
#define IN_DIM 1024
#define H_DIM 1024
#define G4_DIM (4 * H_DIM)

typedef __attribute__((ext_vector_type(16))) __bf16 v16bf;
typedef __attribute__((ext_vector_type(8)))  __bf16 v8bf;
typedef __attribute__((ext_vector_type(8)))  float  v8f;

// ---------------------------------------------------------------- WMMA helper
__device__ inline v8f wmma_bf16(v16bf a, v16bf b, v8f c) {
    // (neg_a, A, neg_b, B, c_mod, C, reuse_a, reuse_b)
    return __builtin_amdgcn_wmma_f32_16x16x32_bf16(false, a, false, b,
                                                   (short)0, c, false, false);
}

// A-matrix 16x32 bf16 per ISA layout:
// lane<16: elems 0..7 = K kc..kc+7,  elems 8..15 = K kc+16..kc+23
// lane>=16: elems 0..7 = K kc+8..+15, elems 8..15 = K kc+24..+31
__device__ inline v16bf load_a_tile(const __bf16* row_ptr, int kc, int lhalf) {
    v8bf lo = *reinterpret_cast<const v8bf*>(row_ptr + kc + lhalf * 8);
    v8bf hi = *reinterpret_cast<const v8bf*>(row_ptr + kc + 16 + lhalf * 8);
    return __builtin_shufflevector(lo, hi, 0, 1, 2, 3, 4, 5, 6, 7,
                                   8, 9, 10, 11, 12, 13, 14, 15);
}

// B-matrix 32x16 bf16: lane holds col N=lane%16; same K order as A tile
// (elems 0..7 = K kc+8*h.., elems 8..15 = K kc+16+8*h..). W stored row-major [4H x K] (row = output gate).
__device__ inline v16bf load_b_tile(const __bf16* w_row_ptr, int kc, int lhalf) {
    v8bf lo = *reinterpret_cast<const v8bf*>(w_row_ptr + kc + lhalf * 8);
    v8bf hi = *reinterpret_cast<const v8bf*>(w_row_ptr + kc + 16 + lhalf * 8);
    return __builtin_shufflevector(lo, hi, 0, 1, 2, 3, 4, 5, 6, 7,
                                   8, 9, 10, 11, 12, 13, 14, 15);
}

__device__ inline float fast_sigmoid(float x) {
    return 1.0f / (1.0f + __expf(-x));
}

// ------------------------------------------------------- Phase 1: convert/init
__global__ void convert_init_kernel(const float* __restrict__ x,
                                    const float* __restrict__ wih,
                                    const float* __restrict__ whh,
                                    const float* __restrict__ h0,
                                    __bf16* __restrict__ xb,
                                    __bf16* __restrict__ wihb,
                                    __bf16* __restrict__ whhb,
                                    __bf16* __restrict__ hbuf0,
                                    unsigned* __restrict__ bar) {
    size_t idx = (size_t)blockIdx.x * blockDim.x + threadIdx.x;
    if (idx < (size_t)T_DIM * B_DIM * IN_DIM) xb[idx]   = (__bf16)x[idx];
    if (idx < (size_t)G4_DIM * IN_DIM)        wihb[idx] = (__bf16)wih[idx];
    if (idx < (size_t)G4_DIM * H_DIM)         whhb[idx] = (__bf16)whh[idx];
    if (idx < (size_t)B_DIM * H_DIM)          hbuf0[idx] = (__bf16)h0[idx];
    if (idx < 2)                              bar[idx] = 0u;  // cnt, gen
}

// ------------------------------------------------- Phase 2: x_proj GEMM (WMMA)
// x_proj stored TRANSPOSED: xproj[t, g, b] = sum_k x[t,b,k]*wih[g,k] + biases.
// Each wave: one t (32 rows) x 32 gate cols, K=1024 in chunks of 32.
__global__ void xproj_gemm_kernel(const __bf16* __restrict__ xb,
                                  const __bf16* __restrict__ wihb,
                                  const float* __restrict__ bias_ih,
                                  const float* __restrict__ bias_hh,
                                  float* __restrict__ xproj) {
    int wid   = (blockIdx.x * blockDim.x + threadIdx.x) >> 5;
    int lane  = threadIdx.x & 31;
    int lrow  = lane & 15;
    int lhalf = lane >> 4;

    int tm = wid >> 7;    // 0..511 == timestep t (M tile of 32 == one t)
    int tn = wid & 127;   // 0..127  (N tiles of 32 over 4H=4096)
    int m0 = tm * 32;
    int n0 = tn * 32;

    const __bf16* arow0 = xb   + (size_t)(m0 + lrow)      * IN_DIM;
    const __bf16* arow1 = xb   + (size_t)(m0 + 16 + lrow) * IN_DIM;
    const __bf16* brow0 = wihb + (size_t)(n0 + lrow)      * IN_DIM;
    const __bf16* brow1 = wihb + (size_t)(n0 + 16 + lrow) * IN_DIM;

    v8f acc00 = {}, acc01 = {}, acc10 = {}, acc11 = {};
#pragma unroll 4
    for (int kc = 0; kc < IN_DIM; kc += 32) {
        v16bf a0 = load_a_tile(arow0, kc, lhalf);
        v16bf a1 = load_a_tile(arow1, kc, lhalf);
        v16bf b0 = load_b_tile(brow0, kc, lhalf);
        v16bf b1 = load_b_tile(brow1, kc, lhalf);
        acc00 = wmma_bf16(a0, b0, acc00);
        acc01 = wmma_bf16(a0, b1, acc01);
        acc10 = wmma_bf16(a1, b0, acc10);
        acc11 = wmma_bf16(a1, b1, acc11);
    }

    int col0 = n0 + lrow;
    int col1 = n0 + 16 + lrow;
    float bb0 = bias_ih[col0] + bias_hh[col0];
    float bb1 = bias_ih[col1] + bias_hh[col1];

    // Transposed store: xproj[(t*4H + g)*B + b]; lane's v=0..7 are 8
    // consecutive b values -> two float4 stores per tile.
    auto store_tile = [&](v8f acc, int col, int b0_, float bb) {
        float* base = xproj + ((size_t)tm * G4_DIM + col) * B_DIM + b0_;
        float4 lo = make_float4(acc[0] + bb, acc[1] + bb, acc[2] + bb, acc[3] + bb);
        float4 hi = make_float4(acc[4] + bb, acc[5] + bb, acc[6] + bb, acc[7] + bb);
        *reinterpret_cast<float4*>(base)     = lo;
        *reinterpret_cast<float4*>(base + 4) = hi;
    };
    int bbase0 = 8 * lhalf;        // rows of first M half
    int bbase1 = 16 + 8 * lhalf;   // rows of second M half
    store_tile(acc00, col0, bbase0, bb0);
    store_tile(acc01, col1, bbase0, bb1);
    store_tile(acc10, col0, bbase1, bb0);
    store_tile(acc11, col1, bbase1, bb1);
}

// --------------------------------------------------------- grid-wide barrier
__device__ inline void grid_barrier(unsigned* cnt, unsigned* gen, unsigned nblk) {
    __builtin_amdgcn_fence(__ATOMIC_RELEASE, "agent");   // flush stores to L2
    __syncthreads();
    if (threadIdx.x == 0) {
        unsigned g = __hip_atomic_load(gen, __ATOMIC_RELAXED,
                                       __HIP_MEMORY_SCOPE_AGENT);
        unsigned a = __hip_atomic_fetch_add(cnt, 1u, __ATOMIC_ACQ_REL,
                                            __HIP_MEMORY_SCOPE_AGENT);
        if (a == nblk - 1u) {
            __hip_atomic_store(cnt, 0u, __ATOMIC_RELAXED,
                               __HIP_MEMORY_SCOPE_AGENT);
            __hip_atomic_store(gen, g + 1u, __ATOMIC_RELEASE,
                               __HIP_MEMORY_SCOPE_AGENT);
        } else {
            while (__hip_atomic_load(gen, __ATOMIC_ACQUIRE,
                                     __HIP_MEMORY_SCOPE_AGENT) == g) {
                __builtin_amdgcn_s_sleep(1);
            }
        }
    }
    __syncthreads();
    __builtin_amdgcn_fence(__ATOMIC_ACQUIRE, "agent");   // invalidate L0 (global_inv)
}

// --------------------------------------- Phase 3: persistent recurrent scan
// 32 blocks x 256 threads = 256 waves = 128 (mhalf,j) slices x 2 K-halves.
// Block: 8 waves = 4 slices x 2 kparts. kpart=1 partials reduced via LDS.
__global__ void __launch_bounds__(256)
recurrent_kernel(const float* __restrict__ xproj,   // [T, 4H, B]
                 const __bf16* __restrict__ whhb,
                 __bf16* __restrict__ hbuf,          // [2][B*H] double buffer
                 const float* __restrict__ h0,
                 const float* __restrict__ c0,
                 const float* __restrict__ rr,
                 float* __restrict__ ys,             // [T,B,H]
                 float* __restrict__ hT,             // [B,H]
                 float* __restrict__ cT,             // [B,H]
                 unsigned* __restrict__ bar) {
    const int wib   = threadIdx.x >> 5;              // wave in block 0..7
    const int wid   = blockIdx.x * 8 + wib;          // 0..255
    const int lane  = threadIdx.x & 31;
    const int lrow  = lane & 15;
    const int lhalf = lane >> 4;

    const int kpart = wid & 1;                       // K half: 0 or 1
    const int combo = wid >> 1;                      // 0..127 slice id
    const int mhalf = combo & 1;                     // which 16 batch rows
    const int j     = combo >> 1;                    // hidden tile 0..63
    const int m0    = mhalf * 16;
    const int jc    = j * 16 + lrow;                 // this lane's hidden col
    const int cl    = wib >> 1;                      // slice within block 0..3

    // LDS reduction buffers: [slice][tile][lane][8]
    __shared__ __align__(32) float red[4][4][32][8];

    const int k_begin = kpart * (H_DIM / 2);
    const int k_end   = k_begin + (H_DIM / 2);

    const float r_lane  = rr[jc];
    const float r_inv   = 1.0f - r_lane;

    // W rows for the 4 gate tiles of hidden columns jc
    const __bf16* w_i = whhb + (size_t)(0 * H_DIM + j * 16 + lrow) * H_DIM;
    const __bf16* w_f = whhb + (size_t)(1 * H_DIM + j * 16 + lrow) * H_DIM;
    const __bf16* w_g = whhb + (size_t)(2 * H_DIM + j * 16 + lrow) * H_DIM;
    const __bf16* w_o = whhb + (size_t)(3 * H_DIM + j * 16 + lrow) * H_DIM;

    // c and h(prev) live in registers (C-tile layout); only kpart==0 uses them
    v8f c_reg = {}, h_reg = {};
    if (kpart == 0) {
#pragma unroll
        for (int v = 0; v < 8; ++v) {
            int b = m0 + 8 * lhalf + v;
            c_reg[v] = c0[(size_t)b * H_DIM + jc];
            h_reg[v] = h0[(size_t)b * H_DIM + jc];
        }
    }

    const unsigned nblk = gridDim.x;
    unsigned* bar_cnt = bar;
    unsigned* bar_gen = bar + 1;

    for (int t = 0; t < T_DIM; ++t) {
        const __bf16* hcur  = hbuf + (size_t)(t & 1) * B_DIM * H_DIM;
        __bf16*       hnext = hbuf + (size_t)((t + 1) & 1) * B_DIM * H_DIM;
        const __bf16* arow  = hcur + (size_t)(m0 + lrow) * H_DIM;

        v8f acc_i = {}, acc_f = {}, acc_g = {}, acc_o = {};
#pragma unroll 4
        for (int kc = k_begin; kc < k_end; kc += 32) {
            v16bf a  = load_a_tile(arow, kc, lhalf);
            v16bf bi = load_b_tile(w_i, kc, lhalf);
            v16bf bf = load_b_tile(w_f, kc, lhalf);
            v16bf bg = load_b_tile(w_g, kc, lhalf);
            v16bf bo = load_b_tile(w_o, kc, lhalf);
            acc_i = wmma_bf16(a, bi, acc_i);
            acc_f = wmma_bf16(a, bf, acc_f);
            acc_g = wmma_bf16(a, bg, acc_g);
            acc_o = wmma_bf16(a, bo, acc_o);
        }

        // K-split reduction through LDS: kpart==1 publishes its partials.
        if (kpart == 1) {
            *reinterpret_cast<v8f*>(&red[cl][0][lane][0]) = acc_i;
            *reinterpret_cast<v8f*>(&red[cl][1][lane][0]) = acc_f;
            *reinterpret_cast<v8f*>(&red[cl][2][lane][0]) = acc_g;
            *reinterpret_cast<v8f*>(&red[cl][3][lane][0]) = acc_o;
        }
        __syncthreads();

        if (kpart == 0) {
            acc_i += *reinterpret_cast<const v8f*>(&red[cl][0][lane][0]);
            acc_f += *reinterpret_cast<const v8f*>(&red[cl][1][lane][0]);
            acc_g += *reinterpret_cast<const v8f*>(&red[cl][2][lane][0]);
            acc_o += *reinterpret_cast<const v8f*>(&red[cl][3][lane][0]);

            // x_proj tiles, transposed layout [T,4H,B]: contiguous in b.
            const float* xp = xproj + (size_t)t * G4_DIM * B_DIM;
            auto load_xp = [&](int gate) -> v8f {
                const float* p = xp + ((size_t)gate * H_DIM + jc) * B_DIM
                               + m0 + 8 * lhalf;
                float4 lo = *reinterpret_cast<const float4*>(p);
                float4 hi = *reinterpret_cast<const float4*>(p + 4);
                v8f r;
                r[0] = lo.x; r[1] = lo.y; r[2] = lo.z; r[3] = lo.w;
                r[4] = hi.x; r[5] = hi.y; r[6] = hi.z; r[7] = hi.w;
                return r;
            };
            v8f xpi = load_xp(0), xpf = load_xp(1), xpg = load_xp(2), xpo = load_xp(3);

#pragma unroll
            for (int v = 0; v < 8; ++v) {
                int b = m0 + 8 * lhalf + v;
                float iv = fast_sigmoid(acc_i[v] + xpi[v]);
                float fv = fast_sigmoid(acc_f[v] + xpf[v]);
                float gv = tanhf(acc_g[v] + xpg[v]);
                float ov = fast_sigmoid(acc_o[v] + xpo[v]);
                float cy = fv * c_reg[v] + iv * gv;
                float hy = ov * tanhf(cy);
                hy = r_lane * h_reg[v] + r_inv * hy;    // retention blend
                c_reg[v] = cy;
                h_reg[v] = hy;
                ys[((size_t)t * B_DIM + b) * H_DIM + jc] = hy;
                hnext[(size_t)b * H_DIM + jc] = (__bf16)hy;
            }
        }

        grid_barrier(bar_cnt, bar_gen, nblk);   // publish h for next step
    }

    if (kpart == 0) {
#pragma unroll
        for (int v = 0; v < 8; ++v) {
            int b = m0 + 8 * lhalf + v;
            hT[(size_t)b * H_DIM + jc] = h_reg[v];
            cT[(size_t)b * H_DIM + jc] = c_reg[v];
        }
    }
}

// ------------------------------------------------------------------- launcher
extern "C" void kernel_launch(void* const* d_in, const int* in_sizes, int n_in,
                              void* d_out, int out_size, void* d_ws, size_t ws_size,
                              hipStream_t stream) {
    const float* x   = (const float*)d_in[0];
    const float* h0  = (const float*)d_in[1];
    const float* c0  = (const float*)d_in[2];
    const float* wih = (const float*)d_in[3];
    const float* whh = (const float*)d_in[4];
    const float* bih = (const float*)d_in[5];
    const float* bhh = (const float*)d_in[6];
    const float* rr  = (const float*)d_in[7];

    char* ws = (char*)d_ws;
    size_t off = 0;
    auto take = [&](size_t bytes) {
        size_t o = off;
        off = (off + bytes + 255) & ~(size_t)255;
        return o;
    };
    unsigned* bar   = (unsigned*)(ws + take(64));
    __bf16*   xb    = (__bf16*)(ws + take((size_t)T_DIM * B_DIM * IN_DIM * 2));
    __bf16*   wihb  = (__bf16*)(ws + take((size_t)G4_DIM * IN_DIM * 2));
    __bf16*   whhb  = (__bf16*)(ws + take((size_t)G4_DIM * H_DIM * 2));
    __bf16*   hbuf  = (__bf16*)(ws + take((size_t)2 * B_DIM * H_DIM * 2));
    float*    xproj = (float*)(ws + take((size_t)T_DIM * B_DIM * G4_DIM * 4));

    float* ys = (float*)d_out;
    float* hT = ys + (size_t)T_DIM * B_DIM * H_DIM;
    float* cT = hT + (size_t)B_DIM * H_DIM;

    // Phase 1: conversions + barrier init (covers max range T*B*IN = 16M elems)
    {
        size_t n = (size_t)T_DIM * B_DIM * IN_DIM;
        int blocks = (int)((n + 255) / 256);
        convert_init_kernel<<<blocks, 256, 0, stream>>>(x, wih, whh, h0,
                                                        xb, wihb, whhb, hbuf, bar);
    }
    // Phase 2: x_proj GEMM — 512*128 = 65536 waves, 8 waves/block
    xproj_gemm_kernel<<<8192, 256, 0, stream>>>(xb, wihb, bih, bhh, xproj);
    // Phase 3: persistent recurrent scan — 32 blocks * 256 threads = 256 waves
    recurrent_kernel<<<32, 256, 0, stream>>>(xproj, whhb, hbuf, h0, c0, rr,
                                             ys, hT, cT, bar);
}
